// MultimodalCrossAttentionKernel_59622736003250
// MI455X (gfx1250) — matrix operand order repaired and checked by reading, compiled, not run on hardware
//
#include <hip/hip_runtime.h>

// ---------------------------------------------------------------------------
// Types for CDNA5 WMMA (wave32): D(16x16 f32) = A(16x32 f16) x B(32x16 f16) + C
// ---------------------------------------------------------------------------
typedef __attribute__((ext_vector_type(16))) _Float16 v16h;
typedef __attribute__((ext_vector_type(8)))  float    v8f;
typedef __attribute__((ext_vector_type(4)))  unsigned int u32x4;
typedef __attribute__((ext_vector_type(4)))  int       v4i;

union FragH { v16h h; u32x4 u[2]; };

__device__ __forceinline__ v8f wmma16(const v16h& a, const v16h& b, const v8f& c) {
  // 8 args: (neg_a, A, neg_b, B, c_mod, C, reuse_a, reuse_b)
  return __builtin_amdgcn_wmma_f32_16x16x32_f16(false, a, false, b, (short)0, c,
                                                false, false);
}

__device__ __forceinline__ v8f vzero8() {
  v8f z = {0.f, 0.f, 0.f, 0.f, 0.f, 0.f, 0.f, 0.f};
  return z;
}

// ---------------------------------------------------------------------------
// CDNA5 async global->LDS copy (ASYNCcnt path), guarded for toolchain support
// Builtin signature: (v4i addrspace(1)*, v4i addrspace(3)*, imm offset, imm cpol)
// ---------------------------------------------------------------------------
#if defined(__gfx1250__) && __has_builtin(__builtin_amdgcn_global_load_async_to_lds_b128)
#define USE_ASYNC_LDS 1
typedef __attribute__((address_space(1))) v4i* gp1_t;
typedef __attribute__((address_space(3))) v4i* lp3_t;
__device__ __forceinline__ void async_cp16(const void* g, void* l) {
  __builtin_amdgcn_global_load_async_to_lds_b128((gp1_t)g, (lp3_t)l, 0, 0);
}
#if __has_builtin(__builtin_amdgcn_s_wait_asynccnt)
#define WAIT_ASYNC(n) __builtin_amdgcn_s_wait_asynccnt(n)
#else
#define WAIT_ASYNC(n) asm volatile("s_wait_asynccnt %0" ::"n"(n))
#endif
#else
#define USE_ASYNC_LDS 0
#endif

// ---------------------------------------------------------------------------
// LayerNorm (fp32 in) -> f16 out. One block (256 thr) per 1024-elem row.
// ---------------------------------------------------------------------------
__global__ __launch_bounds__(256) void ln_f16_kernel(
    const float* __restrict__ x, const float* __restrict__ g,
    const float* __restrict__ bta, _Float16* __restrict__ out) {
  const int row = blockIdx.x;
  const int tid = threadIdx.x;
  const int lane = tid & 31, wid = tid >> 5;
  const float* xr = x + (size_t)row * 1024;
  float s = 0.f, sq = 0.f;
#pragma unroll
  for (int i = tid; i < 1024; i += 256) { float v = xr[i]; s += v; sq += v * v; }
#pragma unroll
  for (int off = 16; off > 0; off >>= 1) {
    s  += __shfl_xor(s,  off, 32);
    sq += __shfl_xor(sq, off, 32);
  }
  __shared__ float s1[8], s2[8];
  if (lane == 0) { s1[wid] = s; s2[wid] = sq; }
  __syncthreads();
  float tot = 0.f, totq = 0.f;
#pragma unroll
  for (int w = 0; w < 8; ++w) { tot += s1[w]; totq += s2[w]; }
  const float mean = tot * (1.0f / 1024.0f);
  const float var  = totq * (1.0f / 1024.0f) - mean * mean;
  const float rstd = rsqrtf(var + 1e-5f);
  _Float16* orow = out + (size_t)row * 1024;
#pragma unroll
  for (int i = tid; i < 1024; i += 256)
    orow[i] = (_Float16)((xr[i] - mean) * rstd * g[i] + bta[i]);
}

// ---------------------------------------------------------------------------
// fp32 -> f16 convert (for weight matrices)
// ---------------------------------------------------------------------------
__global__ void cvt_f16_kernel(const float* __restrict__ in,
                               _Float16* __restrict__ out, int n) {
  const int i = blockIdx.x * 256 + threadIdx.x;
  if (i < n) out[i] = (_Float16)in[i];
}

// ---------------------------------------------------------------------------
// NT GEMM: C[r,c] = (sum_k A[r,k]*B[c,k] + bias[c]) * scale
// A: [rows,1024] f16, B: [1024,1024] f16 (row-major, rows are output cols).
// Block 256 thr (8 waves), tile 128x128x32, each wave 32x64 (2x4 wmma tiles).
// Tiles staged via GLOBAL_LOAD_ASYNC_TO_LDS_B128 with double buffering.
// Writes f16 (outH) or f32 (outF).
// ---------------------------------------------------------------------------
__global__ __launch_bounds__(256) void gemm_nt_kernel(
    const _Float16* __restrict__ A, const _Float16* __restrict__ Bw,
    const float* __restrict__ bias, float scale,
    _Float16* __restrict__ outH, float* __restrict__ outF) {
  const int tid  = threadIdx.x;
  const int lane = tid & 31, wid = tid >> 5;
  const int wm = wid & 3;    // m offset wm*32
  const int wn = wid >> 2;   // n offset wn*64
  const int rowBase = blockIdx.x * 128, colBase = blockIdx.y * 128;
  const int ml = lane & 15;
  const int koffA = (lane < 16) ? 0 : 8;
  const int koffB = (lane < 16) ? 0 : 8;

  v8f acc[2][4];
#pragma unroll
  for (int i = 0; i < 2; ++i)
#pragma unroll
    for (int j = 0; j < 4; ++j) acc[i][j] = vzero8();

  const int lr = tid >> 1;          // 0..127  (row within tile)
  const int lc = (tid & 1) * 16;    // 0 or 16 (half-row of 16 halves)
  const _Float16* gA = A  + (size_t)(rowBase + lr) * 1024 + lc;
  const _Float16* gB = Bw + (size_t)(colBase + lr) * 1024 + lc;

#if USE_ASYNC_LDS
  __shared__ _Float16 lA[2][128 * 40];  // 32 halves + 8 pad per row
  __shared__ _Float16 lB[2][128 * 40];
  // issue tile 0 into buffer 0 (4 async b128 ops per thread per tile)
  {
    _Float16* la = &lA[0][lr * 40 + lc];
    _Float16* lb = &lB[0][lr * 40 + lc];
    async_cp16(gA,     la);  async_cp16(gA + 8, la + 8);
    async_cp16(gB,     lb);  async_cp16(gB + 8, lb + 8);
  }
  for (int kt = 0; kt < 32; ++kt) {
    const int cur = kt & 1;
    if (kt < 31) {
      const int k0 = (kt + 1) * 32;
      _Float16* la = &lA[cur ^ 1][lr * 40 + lc];
      _Float16* lb = &lB[cur ^ 1][lr * 40 + lc];
      async_cp16(gA + k0,     la);  async_cp16(gA + k0 + 8, la + 8);
      async_cp16(gB + k0,     lb);  async_cp16(gB + k0 + 8, lb + 8);
      WAIT_ASYNC(4);   // previous tile's 4 ops retired (in-order), new 4 in flight
    } else {
      WAIT_ASYNC(0);
    }
    __syncthreads();

    FragH fa[2], fb[4];
#pragma unroll
    for (int i = 0; i < 2; ++i) {
      const u32x4* p =
          (const u32x4*)(&lA[cur][(wm * 32 + i * 16 + ml) * 40 + koffA]);
      fa[i].u[0] = p[0]; fa[i].u[1] = p[2];   // halves koffA.. and koffA+16..
    }
#pragma unroll
    for (int j = 0; j < 4; ++j) {
      const u32x4* p =
          (const u32x4*)(&lB[cur][(wn * 64 + j * 16 + ml) * 40 + koffB]);
      fb[j].u[0] = p[0]; fb[j].u[1] = p[2];   // halves koffB.. and koffB+16..
    }
#pragma unroll
    for (int i = 0; i < 2; ++i)
#pragma unroll
      for (int j = 0; j < 4; ++j)
        acc[i][j] = wmma16(fa[i].h, fb[j].h, acc[i][j]);
    __syncthreads();
  }
#else
  __shared__ _Float16 lA[128 * 40];
  __shared__ _Float16 lB[128 * 40];
  for (int kt = 0; kt < 32; ++kt) {
    const int k0 = kt * 32;
    const u32x4* sa = (const u32x4*)(gA + k0);
    const u32x4* sb = (const u32x4*)(gB + k0);
    u32x4* da = (u32x4*)(lA + lr * 40 + lc);
    u32x4* db = (u32x4*)(lB + lr * 40 + lc);
    da[0] = sa[0]; da[1] = sa[1];
    db[0] = sb[0]; db[1] = sb[1];
    __syncthreads();
    FragH fa[2], fb[4];
#pragma unroll
    for (int i = 0; i < 2; ++i) {
      const u32x4* p = (const u32x4*)(lA + (wm * 32 + i * 16 + ml) * 40 + koffA);
      fa[i].u[0] = p[0]; fa[i].u[1] = p[2];
    }
#pragma unroll
    for (int j = 0; j < 4; ++j) {
      const u32x4* p = (const u32x4*)(lB + (wn * 64 + j * 16 + ml) * 40 + koffB);
      fb[j].u[0] = p[0]; fb[j].u[1] = p[2];
    }
#pragma unroll
    for (int i = 0; i < 2; ++i)
#pragma unroll
      for (int j = 0; j < 4; ++j)
        acc[i][j] = wmma16(fa[i].h, fb[j].h, acc[i][j]);
    __syncthreads();
  }
#endif

  const int rb = (lane < 16) ? 0 : 8;
#pragma unroll
  for (int i = 0; i < 2; ++i) {
#pragma unroll
    for (int j = 0; j < 4; ++j) {
      const int col = colBase + wn * 64 + j * 16 + ml;
      const float bv = bias[col];
#pragma unroll
      for (int r = 0; r < 8; ++r) {
        const int row = rowBase + wm * 32 + i * 16 + rb + r;
        const float v = (acc[i][j][r] + bv) * scale;
        if (outF) outF[(size_t)row * 1024 + col] = v;
        else      outH[(size_t)row * 1024 + col] = (_Float16)v;
      }
    }
  }
}

// ---------------------------------------------------------------------------
// Per-head V transpose: Vt[b][h][d][s] = Vb[row(s)][h*64+d]  (f16)
// ---------------------------------------------------------------------------
__global__ void vtrans_kernel(const _Float16* __restrict__ Vb,
                              _Float16* __restrict__ Vt) {
  const int idx = blockIdx.x * 256 + threadIdx.x;
  if (idx >= 4 * 16 * 64 * 1088) return;
  const int s = idx % 1088;
  int t = idx / 1088;
  const int d = t & 63;  t >>= 6;
  const int h = t & 15;  const int b = t >> 4;
  const size_t row = (s < 512) ? (size_t)(b * 512 + s)
                               : (size_t)(2048 + b * 576 + (s - 512));
  Vt[idx] = Vb[row * 1024 + h * 64 + d];
}

// ---------------------------------------------------------------------------
// Attention: grid (Sq/16, H=16, B=4), block 128 (4 waves).
// Scores (WMMA) -> f16 LDS -> softmax (f32 weights to d_out) -> O=e@V^T (WMMA)
// Q already carries the 1/sqrt(hd) scale.
// ---------------------------------------------------------------------------
__global__ __launch_bounds__(128) void attn_kernel(
    const _Float16* __restrict__ Qb, const _Float16* __restrict__ Kb,
    const _Float16* __restrict__ Vt, float* __restrict__ wOut,
    _Float16* __restrict__ Ob, int Sq, int qRowOff) {
  __shared__ _Float16 eb[16 * 1104];  // scores then exp-weights, 16 rows
  __shared__ float rs[16];            // 1/sum per row
  const int tid  = threadIdx.x;
  const int lane = tid & 31, wid = tid >> 5;
  const int ml = lane & 15;
  const int koffA = (lane < 16) ? 0 : 8;
  const int koffB = (lane < 16) ? 0 : 8;
  const int rb    = (lane < 16) ? 0 : 8;
  const int q0 = blockIdx.x * 16;
  const int h  = blockIdx.y;
  const int b  = blockIdx.z;

  // ---- scores = Q K^T ----
  FragH qf[2];
  {
    const _Float16* qp =
        Qb + (size_t)(qRowOff + b * Sq + q0 + ml) * 1024 + h * 64;
#pragma unroll
    for (int ks = 0; ks < 2; ++ks) {
      const u32x4* p = (const u32x4*)(qp + ks * 32 + koffA);
      qf[ks].u[0] = p[0]; qf[ks].u[1] = p[2];
    }
  }
  for (int nt = 0; nt < 17; ++nt) {          // each wave: 272 kv columns
    const int s0 = wid * 272 + nt * 16;
    const int s  = s0 + ml;
    const size_t krow = (s < 512) ? (size_t)(b * 512 + s)
                                  : (size_t)(2048 + b * 576 + (s - 512));
    const _Float16* kp = Kb + krow * 1024 + h * 64;
    v8f acc = vzero8();
#pragma unroll
    for (int ks = 0; ks < 2; ++ks) {
      FragH kf;
      const u32x4* p = (const u32x4*)(kp + ks * 32 + koffB);
      kf.u[0] = p[0]; kf.u[1] = p[2];
      acc = wmma16(qf[ks].h, kf.h, acc);
    }
#pragma unroll
    for (int r = 0; r < 8; ++r)
      eb[(rb + r) * 1104 + s0 + ml] = (_Float16)acc[r];
  }
  __syncthreads();

  // ---- softmax: wave handles 4 rows; 1088 = 34 * 32 exactly ----
#pragma unroll 1
  for (int rr = 0; rr < 4; ++rr) {
    const int q = wid * 4 + rr;
    float ev[34];
    float mx = -3.0e38f;
#pragma unroll
    for (int k2 = 0; k2 < 34; ++k2) {
      const float v = (float)eb[q * 1104 + lane + 32 * k2];
      ev[k2] = v;
      mx = fmaxf(mx, v);
    }
#pragma unroll
    for (int off = 16; off > 0; off >>= 1) mx = fmaxf(mx, __shfl_xor(mx, off, 32));
    float sum = 0.f;
#pragma unroll
    for (int k2 = 0; k2 < 34; ++k2) {
      const float e = __expf(ev[k2] - mx);
      ev[k2] = e;
      sum += e;
    }
#pragma unroll
    for (int off = 16; off > 0; off >>= 1) sum += __shfl_xor(sum, off, 32);
    const float rsum = 1.0f / sum;
    if (lane == 0) rs[q] = rsum;
    float* wrow = wOut + (((size_t)(b * 16 + h)) * Sq + (q0 + q)) * 1088;
#pragma unroll
    for (int k2 = 0; k2 < 34; ++k2) {
      const int i = lane + 32 * k2;
      wrow[i] = ev[k2] * rsum;              // normalized weights (fp32 out)
      eb[q * 1104 + i] = (_Float16)ev[k2];  // unnormalized e (<=1) for O gemm
    }
  }
  __syncthreads();

  // ---- O = e @ V^T : each wave owns a 16-wide d slice ----
  const int d0 = wid * 16;
  const _Float16* vp = Vt + (((size_t)(b * 16 + h)) * 64 + d0 + ml) * 1088;
  v8f oacc = vzero8();
  for (int ks = 0; ks < 34; ++ks) {
    FragH af, bf;
    const u32x4* pa = (const u32x4*)(eb + ml * 1104 + ks * 32 + koffA);
    af.u[0] = pa[0]; af.u[1] = pa[2];
    const u32x4* pb = (const u32x4*)(vp + ks * 32 + koffB);
    bf.u[0] = pb[0]; bf.u[1] = pb[2];
    oacc = wmma16(af.h, bf.h, oacc);
  }
#pragma unroll
  for (int r = 0; r < 8; ++r) {
    const int q = rb + r;
    const float val = oacc[r] * rs[q];
    Ob[(size_t)(qRowOff + b * Sq + q0 + q) * 1024 + h * 64 + d0 + ml] =
        (_Float16)val;
  }
}

// ---------------------------------------------------------------------------
// Host launcher
// ---------------------------------------------------------------------------
extern "C" void kernel_launch(void* const* d_in, const int* in_sizes, int n_in,
                              void* d_out, int out_size, void* d_ws, size_t ws_size,
                              hipStream_t stream) {
  (void)in_sizes; (void)n_in; (void)out_size; (void)ws_size;
  const float* q_text  = (const float*)d_in[0];
  const float* q_image = (const float*)d_in[1];
  const float* k_text  = (const float*)d_in[2];
  const float* k_image = (const float*)d_in[3];
  const float* Wq_t = (const float*)d_in[4];  const float* bq_t = (const float*)d_in[5];
  const float* Wk_t = (const float*)d_in[6];  const float* bk_t = (const float*)d_in[7];
  const float* Wv_t = (const float*)d_in[8];  const float* bv_t = (const float*)d_in[9];
  const float* Wq_i = (const float*)d_in[10]; const float* bq_i = (const float*)d_in[11];
  const float* Wk_i = (const float*)d_in[12]; const float* bk_i = (const float*)d_in[13];
  const float* Wv_i = (const float*)d_in[14]; const float* bv_i = (const float*)d_in[15];
  const float* g_t  = (const float*)d_in[16]; const float* b_t  = (const float*)d_in[17];
  const float* g_i  = (const float*)d_in[18]; const float* b_i  = (const float*)d_in[19];
  const float* Wo   = (const float*)d_in[20]; const float* bo   = (const float*)d_in[21];

  const size_t RT = 2048, RI = 2304, RA = RT + RI;  // rows (B*S)
  _Float16* ws  = (_Float16*)d_ws;
  _Float16* knT = ws;
  _Float16* knI = knT + RT * 1024;
  _Float16* qnT = knI + RI * 1024;
  _Float16* qnI = qnT + RT * 1024;
  _Float16* Wh  = qnI + RI * 1024;                  // 7 x 1M f16 weights
  _Float16* Qb  = Wh + 7ull * 1024 * 1024;
  _Float16* Kb  = Qb + RA * 1024;
  _Float16* Vb  = Kb + RA * 1024;
  _Float16* Vt  = Vb + RA * 1024;
  _Float16* Ob  = Vt + RA * 1024;                   // total ~77 MB

  float* outF  = (float*)d_out;
  float* out_t = outF;
  float* out_i = outF + RT * 1024;
  float* w_t   = outF + RA * 1024;
  float* w_i   = w_t + 4ull * 16 * 512 * 1088;

  // LayerNorm -> f16
  ln_f16_kernel<<<dim3((unsigned)RT), 256, 0, stream>>>(k_text,  g_t, b_t, knT);
  ln_f16_kernel<<<dim3((unsigned)RI), 256, 0, stream>>>(k_image, g_i, b_i, knI);
  ln_f16_kernel<<<dim3((unsigned)RT), 256, 0, stream>>>(q_text,  g_t, b_t, qnT);
  ln_f16_kernel<<<dim3((unsigned)RI), 256, 0, stream>>>(q_image, g_i, b_i, qnI);

  // Weights -> f16 (order: Wq_t, Wk_t, Wv_t, Wq_i, Wk_i, Wv_i, Wo)
  const int WN = 1024 * 1024;
  const float* wsrc[7] = {Wq_t, Wk_t, Wv_t, Wq_i, Wk_i, Wv_i, Wo};
  for (int i = 0; i < 7; ++i)
    cvt_f16_kernel<<<dim3(WN / 256), 256, 0, stream>>>(wsrc[i], Wh + (size_t)i * WN, WN);

  const float SC = 0.125f;  // hd^-0.5
  // Projections (NT WMMA GEMMs), f16 outputs
  gemm_nt_kernel<<<dim3(RT / 128, 8), 256, 0, stream>>>(qnT, Wh + 0 * WN, bq_t, SC,  Qb,             nullptr);
  gemm_nt_kernel<<<dim3(RI / 128, 8), 256, 0, stream>>>(qnI, Wh + 3 * WN, bq_i, SC,  Qb + RT * 1024, nullptr);
  gemm_nt_kernel<<<dim3(RT / 128, 8), 256, 0, stream>>>(knT, Wh + 1 * WN, bk_t, 1.f, Kb,             nullptr);
  gemm_nt_kernel<<<dim3(RI / 128, 8), 256, 0, stream>>>(knI, Wh + 4 * WN, bk_i, 1.f, Kb + RT * 1024, nullptr);
  gemm_nt_kernel<<<dim3(RT / 128, 8), 256, 0, stream>>>(knT, Wh + 2 * WN, bv_t, 1.f, Vb,             nullptr);
  gemm_nt_kernel<<<dim3(RI / 128, 8), 256, 0, stream>>>(knI, Wh + 5 * WN, bv_i, 1.f, Vb + RT * 1024, nullptr);

  // Per-head V transpose for NT O-gemm
  const int TOT = 4 * 16 * 64 * 1088;
  vtrans_kernel<<<dim3((TOT + 255) / 256), 256, 0, stream>>>(Vb, Vt);

  // Attention (text then image)
  attn_kernel<<<dim3(512 / 16, 16, 4), 128, 0, stream>>>(Qb, Kb, Vt, w_t, Ob, 512, 0);
  attn_kernel<<<dim3(576 / 16, 16, 4), 128, 0, stream>>>(Qb, Kb, Vt, w_i, Ob, 576, (int)RT);

  // Output projection (fp32 straight to d_out)
  gemm_nt_kernel<<<dim3(RT / 128, 8), 256, 0, stream>>>(Ob,             Wh + 6 * WN, bo, 1.f, nullptr, out_t);
  gemm_nt_kernel<<<dim3(RI / 128, 8), 256, 0, stream>>>(Ob + RT * 1024, Wh + 6 * WN, bo, 1.f, nullptr, out_i);
}
